// BatchGAT_26482768347246
// MI455X (gfx1250) — matrix operand order repaired and checked by reading, not compiled, not run on hardware
//
#include <hip/hip_runtime.h>
#include <hip/hip_bf16.h>

typedef __attribute__((ext_vector_type(16))) _Float16 v16h;
typedef __attribute__((ext_vector_type(8)))  _Float16 h8;
typedef __attribute__((ext_vector_type(8)))  float    v8f;

#define BB 32
#define NN 512
#define MTOT (BB * NN)
#define NEGINF (-1e20f)

// ---------------------------------------------------------------------------
// WMMA fragment builders, contiguous k order (k = 16g + e) used for both A and B:
// A (16x32 f16): lane l (g=l>>4, m=l&15) holds A[m][k+16g..+15]
//                -> 32 contiguous bytes (2x b128).
__device__ inline v16h frag_a(const _Float16* __restrict__ p0) {
  union { v16h v; h8 h[2]; } u;
  u.h[0] = *(const h8*)p0;
  u.h[1] = *(const h8*)(p0 + 8);
  return u.v;
}
// B (32x16 f16) from a TRANSPOSED operand: lane l (g=l>>4, n=l&15) needs
// B[k+16g+e][n] = BT[n][k+16g+e], e=0..15 -> 32 contiguous bytes (2x b128).
__device__ inline v16h frag_bT(const _Float16* __restrict__ p0) {
  return *(const v16h*)p0;
}

// ---------------------------------------------------------------------------
__global__ void cvt_f32_f16(const float* __restrict__ src,
                            _Float16* __restrict__ dst, int n) {
  int i = blockIdx.x * blockDim.x + threadIdx.x;
  if (i < n) dst[i] = (_Float16)src[i];
}

__global__ void zero_f32(float* __restrict__ p, int n) {
  int i = blockIdx.x * blockDim.x + threadIdx.x;
  if (i < n) p[i] = 0.f;
}

// W (K x Nc) f32  ->  WT (Nc x K) f16
__global__ void cvt_transpose(const float* __restrict__ src,
                              _Float16* __restrict__ dst, int K, int Nc) {
  int idx = blockIdx.x * blockDim.x + threadIdx.x;
  if (idx < K * Nc) {
    int k = idx / Nc, n = idx % Nc;
    dst[n * K + k] = (_Float16)src[idx];
  }
}

// [a_src | a_dst] (64 x 16) -> aT (16 x 64) f16
__global__ void combine_aT(const float* __restrict__ asrc,
                           const float* __restrict__ adst,
                           _Float16* __restrict__ aT) {
  int idx = blockIdx.x * blockDim.x + threadIdx.x;
  if (idx < 16 * 64) {
    int n = idx >> 6, k = idx & 63;
    float v = (n < 8) ? asrc[k * 8 + n] : adst[k * 8 + (n - 8)];
    aT[n * 64 + k] = (_Float16)v;
  }
}

// ---------------------------------------------------------------------------
// C(M x 64) = A(M x K) @ B(K x 64); B given transposed (64 x K).
// Writes C row-major AND CT in batch-blocked transposed layout
// CT[(b*64 + col)*512 + j]  (row = b*512 + j) for the attention B-operand.
template <int K>
__global__ void gemm_nt(const _Float16* __restrict__ A,
                        const _Float16* __restrict__ BT,
                        _Float16* __restrict__ C,
                        _Float16* __restrict__ CT,
                        int total_waves) {
  int wave = (blockIdx.x * blockDim.x + threadIdx.x) >> 5;
  if (wave >= total_waves) return;
  int lane = threadIdx.x & 31;
  int m0 = (wave >> 2) * 16;          // 4 n-tiles (Nc = 64)
  int n0 = (wave & 3) * 16;
  int g = lane >> 4, mr = lane & 15, nc = lane & 15;

  const _Float16* arow = A + (size_t)(m0 + mr) * K + 16 * g;
  const _Float16* brow = BT + (size_t)(n0 + nc) * K + 16 * g;

  v8f acc = {};
#pragma unroll
  for (int k = 0; k < K; k += 32) {
    __builtin_prefetch(arow + k + 128, 0, 1);        // global_prefetch_b8
    v16h a = frag_a(arow + k);
    v16h b = frag_bT(brow + k);
    acc = __builtin_amdgcn_wmma_f32_16x16x32_f16(false, a, false, b,
                                                 (short)0, acc, false, false);
  }
#pragma unroll
  for (int i = 0; i < 8; ++i) {
    int row = m0 + i + 8 * g;
    _Float16 v = (_Float16)acc[i];
    C[(size_t)row * 64 + n0 + nc] = v;
    CT[((size_t)(row >> 9) * 64 + n0 + nc) * NN + (row & (NN - 1))] = v;
  }
}

// ---------------------------------------------------------------------------
// sd(M x 16) = hp(M x 64) @ [a_src|a_dst](64 x 16), via transposed aT(16x64).
__global__ void proj_sd(const _Float16* __restrict__ hp,
                        const _Float16* __restrict__ aT,
                        float* __restrict__ sd, int total_waves) {
  int wave = (blockIdx.x * blockDim.x + threadIdx.x) >> 5;
  if (wave >= total_waves) return;
  int lane = threadIdx.x & 31;
  int m0 = wave * 16;
  int g = lane >> 4, mr = lane & 15, nc = lane & 15;

  const _Float16* arow = hp + (size_t)(m0 + mr) * 64 + 16 * g;
  const _Float16* brow = aT + (size_t)nc * 64 + 16 * g;

  v8f acc = {};
#pragma unroll
  for (int k = 0; k < 64; k += 32) {
    v16h a = frag_a(arow + k);
    v16h b = frag_bT(brow + k);
    acc = __builtin_amdgcn_wmma_f32_16x16x32_f16(false, a, false, b,
                                                 (short)0, acc, false, false);
  }
#pragma unroll
  for (int i = 0; i < 8; ++i)
    sd[(size_t)(m0 + i + 8 * g) * 16 + nc] = acc[i];
}

// ---------------------------------------------------------------------------
// Fused attention: one block = (batch b, head h, 16-row tile), 128 thr = 4 waves.
// Phase 1: masked leaky-relu scores + row softmax in LDS (f32 -> f16 P tile).
// Phase 2: out(16x64) = P(16x512) @ hp_b(512x64) with hp given transposed
// (hpT[b][f][j]) so B fragments are contiguous 32B loads.
__global__ void attn_fused(const float* __restrict__ adj,
                           const float* __restrict__ sd,
                           const _Float16* __restrict__ hpT,
                           _Float16* __restrict__ x1out,
                           float* __restrict__ outmean,
                           int layer) {
  __shared__ float eS[16 * NN];                       // 32 KB
  __shared__ __align__(32) _Float16 pS[16 * NN];      // 16 KB
  __shared__ float red[128];
  __shared__ float rowstat[16];

  const int rt = blockIdx.x;            // row tile 0..31
  const int h  = blockIdx.y;            // head 0..7
  const int b  = blockIdx.z;            // batch 0..31
  const int t  = threadIdx.x;
  const int r  = t >> 3, q = t & 7;     // 16 rows x 8 lanes
  const int i  = rt * 16 + r;

  const float  s_i    = sd[((size_t)b * NN + i) * 16 + h];
  const float* adjrow = adj + ((size_t)b * NN + i) * NN;

  float mx = -3.0e38f;
  for (int j = q; j < NN; j += 8) {
    float d_j = sd[((size_t)b * NN + j) * 16 + 8 + h];
    float v = s_i + d_j;
    v = (v >= 0.f) ? v : 0.2f * v;            // leaky relu
    float av = adjrow[j];
    if (j == i) av = 1.f;                     // forced self loop
    if (av == 0.f || v == 0.f) v = NEGINF;    // reference masking semantics
    eS[r * NN + j] = v;
    mx = fmaxf(mx, v);
  }
  red[t] = mx;
  __syncthreads();
  if (q == 0) {
    float m = red[t];
#pragma unroll
    for (int u = 1; u < 8; ++u) m = fmaxf(m, red[t + u]);
    rowstat[r] = m;
  }
  __syncthreads();
  const float m = rowstat[r];
  float sum = 0.f;
  for (int j = q; j < NN; j += 8) {
    float p = __expf(eS[r * NN + j] - m);
    eS[r * NN + j] = p;
    sum += p;
  }
  __syncthreads();
  red[t] = sum;
  __syncthreads();
  if (q == 0) {
    float ssum = red[t];
#pragma unroll
    for (int u = 1; u < 8; ++u) ssum += red[t + u];
    rowstat[r] = ssum;
  }
  __syncthreads();
  const float inv = 1.f / rowstat[r];
  for (int j = q; j < NN; j += 8)
    pS[r * NN + j] = (_Float16)(eS[r * NN + j] * inv);
  __syncthreads();

  // WMMA: wave wv owns output columns [wv*16, wv*16+16)
  const int lane = t & 31;
  const int n0 = (t >> 5) * 16;
  const int g = lane >> 4, mr = lane & 15, nc = lane & 15;
  const _Float16* brow = hpT + ((size_t)b * 64 + n0 + nc) * NN + 16 * g;

  v8f acc = {};
#pragma unroll 4
  for (int k = 0; k < NN; k += 32) {
    v16h a = frag_a(&pS[mr * NN + k + 16 * g]);      // 2x ds_load_b128
    v16h bb = frag_bT(brow + k);                     // 2x global_load_b128
    acc = __builtin_amdgcn_wmma_f32_16x16x32_f16(false, a, false, bb,
                                                 (short)0, acc, false, false);
  }

  if (layer == 0) {
#pragma unroll
    for (int ii = 0; ii < 8; ++ii) {
      int row = rt * 16 + ii + 8 * g;
      float v = acc[ii];
      v = (v > 0.f) ? v : (__expf(v) - 1.f);         // ELU
      x1out[((size_t)b * NN + row) * 512 + h * 64 + n0 + nc] = (_Float16)v;
    }
  } else {
#pragma unroll
    for (int ii = 0; ii < 8; ++ii) {
      int row = rt * 16 + ii + 8 * g;
      atomicAdd(&outmean[((size_t)b * NN + row) * 64 + n0 + nc],
                acc[ii] * 0.125f);                   // mean over 8 heads
    }
  }
}

// ---------------------------------------------------------------------------
extern "C" void kernel_launch(void* const* d_in, const int* in_sizes, int n_in,
                              void* d_out, int out_size, void* d_ws, size_t ws_size,
                              hipStream_t stream) {
  const float* x   = (const float*)d_in[0];
  const float* adj = (const float*)d_in[1];
  const float* W0  = (const float*)d_in[4];
  const float* as0 = (const float*)d_in[5];
  const float* ad0 = (const float*)d_in[6];
  const float* W1  = (const float*)d_in[7];
  const float* as1 = (const float*)d_in[8];
  const float* ad1 = (const float*)d_in[9];
  float* out = (float*)d_out;

  char* ws = (char*)d_ws;
  size_t off = 0;
  auto alloc = [&](size_t bytes) -> void* {
    void* p = ws + off;
    off = (off + bytes + 255) & ~(size_t)255;
    return p;
  };
  _Float16* xh   = (_Float16*)alloc((size_t)MTOT * 64 * 2);
  _Float16* hp0h = (_Float16*)alloc((size_t)MTOT * 64 * 2);
  _Float16* hpT0 = (_Float16*)alloc((size_t)MTOT * 64 * 2);
  _Float16* x1h  = (_Float16*)alloc((size_t)MTOT * 512 * 2);
  _Float16* hp1h = (_Float16*)alloc((size_t)MTOT * 64 * 2);
  _Float16* hpT1 = (_Float16*)alloc((size_t)MTOT * 64 * 2);
  float*    sd0  = (float*)alloc((size_t)MTOT * 16 * 4);
  float*    sd1  = (float*)alloc((size_t)MTOT * 16 * 4);
  _Float16* w0T  = (_Float16*)alloc(64 * 64 * 2);
  _Float16* w1T  = (_Float16*)alloc(64 * 512 * 2);
  _Float16* aT0  = (_Float16*)alloc(16 * 64 * 2);
  _Float16* aT1  = (_Float16*)alloc(16 * 64 * 2);

  cvt_f32_f16<<<(MTOT * 64 + 255) / 256, 256, 0, stream>>>(x, xh, MTOT * 64);
  cvt_transpose<<<(64 * 64 + 255) / 256, 256, 0, stream>>>(W0, w0T, 64, 64);
  cvt_transpose<<<(512 * 64 + 255) / 256, 256, 0, stream>>>(W1, w1T, 512, 64);
  combine_aT<<<(16 * 64 + 255) / 256, 256, 0, stream>>>(as0, ad0, aT0);
  combine_aT<<<(16 * 64 + 255) / 256, 256, 0, stream>>>(as1, ad1, aT1);

  dim3 agrid(NN / 16, 8, BB);
  const int gwaves = (MTOT / 16) * 4;   // 16x16 tiles over M x 64
  const int pwaves = MTOT / 16;

  // ---- layer 0 ----
  gemm_nt<64><<<(gwaves + 7) / 8, 256, 0, stream>>>(xh, w0T, hp0h, hpT0, gwaves);
  proj_sd<<<(pwaves + 7) / 8, 256, 0, stream>>>(hp0h, aT0, sd0, pwaves);
  attn_fused<<<agrid, 128, 0, stream>>>(adj, sd0, hpT0, x1h, out, 0);

  // ---- layer 1 ----
  gemm_nt<512><<<(gwaves + 7) / 8, 256, 0, stream>>>(x1h, w1T, hp1h, hpT1, gwaves);
  proj_sd<<<(pwaves + 7) / 8, 256, 0, stream>>>(hp1h, aT1, sd1, pwaves);
  zero_f32<<<(MTOT * 64 + 255) / 256, 256, 0, stream>>>(out, MTOT * 64);
  attn_fused<<<agrid, 128, 0, stream>>>(adj, sd1, hpT1, (_Float16*)nullptr,
                                        out, 1);
}
